// Attention_72722386256005
// MI455X (gfx1250) — matrix operand order repaired and checked by reading, not compiled, not run on hardware
//
#include <hip/hip_runtime.h>

typedef _Float16 half_t;
typedef __attribute__((ext_vector_type(8)))  _Float16 v8h;
typedef __attribute__((ext_vector_type(16))) _Float16 v16h;
typedef __attribute__((ext_vector_type(8)))  float    v8f;
typedef __attribute__((ext_vector_type(4)))  int      v4i;

static constexpr int BATCH = 4;
static constexpr int SEQ   = 2048;
static constexpr int CH    = 1024;   // C
static constexpr int NH    = 16;     // heads
static constexpr int HD    = 64;     // head dim

union ABu { v16h v; v8h h[2]; };

// 16x32 fp16 A fragment (row-major source, contiguous K).
// Layout (ISA 7.12.2): lanes 0-15: V0-3 = K 0..7, V4-7 = K 16..23 (row M=lane);
// lanes 16-31: V0-3 = K 8..15, V4-7 = K 24..31 (row M=lane-16).
static __device__ inline v16h load_A16x32(const half_t* base, int ld, int row0, int kb, int lane) {
  const half_t* p = base + (size_t)(row0 + (lane & 15)) * ld + kb + ((lane >> 4) << 3);
  ABu u;
  u.h[0] = *(const v8h*)(p);
  u.h[1] = *(const v8h*)(p + 16);
  return u.v;
}

// 32x16 fp16 B fragment. Source is "N-major" (each output column n is a
// contiguous-K row of `base`, stride ld), which matches W ([N,K] row-major),
// K-tiles in LDS (key rows x dims) and transposed-V in LDS (dim rows x keys).
// Layout: lanes 0-15 hold K 0..7,16..23 (col N=lane), lanes 16-31 hold K 8..15,24..31.
static __device__ inline v16h load_B32x16(const half_t* base, int ld, int col0, int kb, int lane) {
  const half_t* p = base + (size_t)(col0 + (lane & 15)) * ld + kb + ((lane >> 4) << 3);
  ABu u;
  u.h[0] = *(const v8h*)(p);
  u.h[1] = *(const v8h*)(p + 16);
  return u.v;
}

static __device__ inline v8f wmma_f32_f16(v16h a, v16h b, v8f c) {
  return __builtin_amdgcn_wmma_f32_16x16x32_f16(false, a, false, b, (short)0, c, false, false);
}

static __device__ inline v8f vzero8f() {
  v8f z;
#pragma unroll
  for (int i = 0; i < 8; ++i) z[i] = 0.0f;
  return z;
}

// ---------------------------------------------------------------------------
// Kernel 1: QKV = X @ Wqkv^T, scattered into [3][B][H][T][D] fp16 workspace.
// One wave -> 32x64 output tile. 8 waves per block.
// ---------------------------------------------------------------------------
__global__ __launch_bounds__(256) void qkv_gemm_kernel(const half_t* __restrict__ X,
                                                       const half_t* __restrict__ W,
                                                       half_t* __restrict__ QKV) {
  const int lane = threadIdx.x & 31;
  const int waveId = blockIdx.x * 8 + (threadIdx.x >> 5);
  const int M_TILES = (BATCH * SEQ) / 32;       // 256
  const int mi = waveId % M_TILES;
  const int ni = waveId / M_TILES;              // 0..47
  const int mr = mi * 32;
  const int nb = ni * 64;

  v8f acc[2][4];
#pragma unroll
  for (int i = 0; i < 2; ++i)
#pragma unroll
    for (int j = 0; j < 4; ++j) acc[i][j] = vzero8f();

  for (int kb = 0; kb < CH; kb += 32) {
    v16h a0 = load_A16x32(X, CH, mr,      kb, lane);
    v16h a1 = load_A16x32(X, CH, mr + 16, kb, lane);
    v16h b0 = load_B32x16(W, CH, nb,      kb, lane);
    v16h b1 = load_B32x16(W, CH, nb + 16, kb, lane);
    v16h b2 = load_B32x16(W, CH, nb + 32, kb, lane);
    v16h b3 = load_B32x16(W, CH, nb + 48, kb, lane);
    acc[0][0] = wmma_f32_f16(a0, b0, acc[0][0]);
    acc[0][1] = wmma_f32_f16(a0, b1, acc[0][1]);
    acc[0][2] = wmma_f32_f16(a0, b2, acc[0][2]);
    acc[0][3] = wmma_f32_f16(a0, b3, acc[0][3]);
    acc[1][0] = wmma_f32_f16(a1, b0, acc[1][0]);
    acc[1][1] = wmma_f32_f16(a1, b1, acc[1][1]);
    acc[1][2] = wmma_f32_f16(a1, b2, acc[1][2]);
    acc[1][3] = wmma_f32_f16(a1, b3, acc[1][3]);
  }

  // C layout: VGPR r -> row = r + 8*(lane/16), col = lane%16.
  const int rhalf = (lane >> 4) << 3;
#pragma unroll
  for (int i = 0; i < 2; ++i) {
#pragma unroll
    for (int j = 0; j < 4; ++j) {
      const int col = nb + 16 * j + (lane & 15);     // 0..3071
      const int p = col >> 10;                        // 0=q,1=k,2=v
      const int rem = col & 1023;
      const int h = rem >> 6;
      const int d = rem & 63;
#pragma unroll
      for (int r = 0; r < 8; ++r) {
        const int row = mr + 16 * i + r + rhalf;      // 0..8191
        const int bb = row >> 11;
        const int t = row & (SEQ - 1);
        const size_t idx = ((((size_t)p * BATCH + bb) * NH + h) * SEQ + t) * HD + d;
        QKV[idx] = (half_t)acc[i][j][r];
      }
    }
  }
}

// ---------------------------------------------------------------------------
// Kernel 2: causal flash attention. Block = 8 waves = 128 query rows of one
// (b,h). K tile staged via async global->LDS DMA (ASYNCcnt); V staged with a
// transpose so P*V B-fragments are contiguous ds_load_b128.
// ---------------------------------------------------------------------------
__global__ __launch_bounds__(256) void flash_attn_kernel(const half_t* __restrict__ QKV,
                                                         half_t* __restrict__ AT) {
  __shared__ half_t Klds[32 * 72];        // [key][dim], stride 72 halves (144B)
  __shared__ half_t Vt[64 * 40];          // [dim][key], stride 40 halves (80B)
  __shared__ half_t Plds[8][16 * 48];     // per-wave P relayout, stride 48 halves

  const int tid = threadIdx.x;
  const int lane = tid & 31;
  const int w = tid >> 5;
  const int qblocks = SEQ / 128;          // 16
  const int bh = blockIdx.x / qblocks;
  const int qb = blockIdx.x % qblocks;
  const int b = bh >> 4;
  const int h = bh & 15;

  const size_t headsz = (size_t)SEQ * HD;
  const half_t* Qp = QKV + ((size_t)(0 * BATCH + b) * NH + h) * headsz;
  const half_t* Kp = QKV + ((size_t)(1 * BATCH + b) * NH + h) * headsz;
  const half_t* Vp = QKV + ((size_t)(2 * BATCH + b) * NH + h) * headsz;

  const int wgq = qb * 128;
  const int qbase = wgq + w * 16;

  const v16h aQ0 = load_A16x32(Qp, HD, qbase, 0, lane);
  const v16h aQ1 = load_A16x32(Qp, HD, qbase, 32, lane);

  v8f o[4];
#pragma unroll
  for (int t = 0; t < 4; ++t) o[t] = vzero8f();
  float m[8], l[8];
#pragma unroll
  for (int r = 0; r < 8; ++r) { m[r] = -__builtin_inff(); l[r] = 0.0f; }

  const float scale = 0.125f;             // 1/sqrt(64)
  const int nblk = (wgq + 128) >> 5;      // 32-key blocks needed by this WG
  const int stg_k = tid >> 3;             // 0..31 key
  const int stg_d = (tid & 7) << 3;       // 0,8,..,56 dim

  for (int blk = 0; blk < nblk; ++blk) {
    const int s0 = blk << 5;
    __syncthreads();                      // previous block's LDS reads done
    {
      // K tile: async DMA global -> LDS (no VGPR round trip, ASYNCcnt).
      const half_t* gK = Kp + (size_t)(s0 + stg_k) * HD + stg_d;
      __builtin_amdgcn_global_load_async_to_lds_b128(
          (__attribute__((address_space(1))) v4i*)gK,
          (__attribute__((address_space(3))) v4i*)&Klds[stg_k * 72 + stg_d],
          0, 0);
      // V tile: load + transpose into LDS (dim-major rows of keys).
      v8h vv = *(const v8h*)(Vp + (size_t)(s0 + stg_k) * HD + stg_d);
#pragma unroll
      for (int e = 0; e < 8; ++e) Vt[(stg_d + e) * 40 + stg_k] = vv[e];
      asm volatile("s_wait_asynccnt 0x0" ::: "memory");
    }
    __syncthreads();
    if (s0 > qbase + 15) continue;        // whole block above our rows (causal)

    // S = Q K^T : two 16x16 tiles (keys 0..15, 16..31), K-dim = 64 head dims.
    v8f S0 = vzero8f();
    S0 = wmma_f32_f16(aQ0, load_B32x16(Klds, 72, 0, 0, lane), S0);
    S0 = wmma_f32_f16(aQ1, load_B32x16(Klds, 72, 0, 32, lane), S0);
    v8f S1 = vzero8f();
    S1 = wmma_f32_f16(aQ0, load_B32x16(Klds, 72, 16, 0, lane), S1);
    S1 = wmma_f32_f16(aQ1, load_B32x16(Klds, 72, 16, 32, lane), S1);

    const int colj = s0 + (lane & 15);
    const int rowb = qbase + ((lane >> 4) << 3);
#pragma unroll
    for (int r = 0; r < 8; ++r) {
      const int i = rowb + r;
      float s0v = (colj <= i) ? S0[r] * scale : -__builtin_inff();
      float s1v = (colj + 16 <= i) ? S1[r] * scale : -__builtin_inff();
      float mx = fmaxf(s0v, s1v);
      mx = fmaxf(mx, __shfl_xor(mx, 1, 32));
      mx = fmaxf(mx, __shfl_xor(mx, 2, 32));
      mx = fmaxf(mx, __shfl_xor(mx, 4, 32));
      mx = fmaxf(mx, __shfl_xor(mx, 8, 32));
      const float nm = fmaxf(m[r], mx);
      const float alpha = __expf(m[r] - nm);
      const float p0 = __expf(s0v - nm);
      const float p1 = __expf(s1v - nm);
      float rs = p0 + p1;
      rs += __shfl_xor(rs, 1, 32);
      rs += __shfl_xor(rs, 2, 32);
      rs += __shfl_xor(rs, 4, 32);
      rs += __shfl_xor(rs, 8, 32);
      l[r] = l[r] * alpha + rs;
      m[r] = nm;
      o[0][r] *= alpha; o[1][r] *= alpha; o[2][r] *= alpha; o[3][r] *= alpha;
      const int rl = ((lane >> 4) << 3) + r;
      Plds[w][rl * 48 + (lane & 15)] = (half_t)p0;
      Plds[w][rl * 48 + 16 + (lane & 15)] = (half_t)p1;
    }

    // O += P V : A = P (16 rows x 32 keys), B = Vt (keys as K, dims as N).
    const v16h aP = load_A16x32(&Plds[w][0], 48, 0, 0, lane);
#pragma unroll
    for (int t = 0; t < 4; ++t) {
      v16h bV = load_B32x16(Vt, 40, 16 * t, 0, lane);
      o[t] = wmma_f32_f16(aP, bV, o[t]);
    }
  }

  // Normalize and write attn (B,T,C) fp16.
#pragma unroll
  for (int r = 0; r < 8; ++r) {
    const float inv = 1.0f / l[r];
    const int row = qbase + ((lane >> 4) << 3) + r;
    const size_t base = ((size_t)b * SEQ + row) * CH + h * HD + (lane & 15);
#pragma unroll
    for (int t = 0; t < 4; ++t) AT[base + 16 * t] = (half_t)(o[t][r] * inv);
  }
}

// ---------------------------------------------------------------------------
// Kernel 3: out = residual + attn @ Wo^T. Same tiling as kernel 1.
// ---------------------------------------------------------------------------
__global__ __launch_bounds__(256) void out_proj_kernel(const half_t* __restrict__ A,
                                                       const half_t* __restrict__ W,
                                                       const half_t* __restrict__ resid,
                                                       half_t* __restrict__ out) {
  const int lane = threadIdx.x & 31;
  const int waveId = blockIdx.x * 8 + (threadIdx.x >> 5);
  const int M_TILES = (BATCH * SEQ) / 32;       // 256
  const int mi = waveId % M_TILES;
  const int ni = waveId / M_TILES;              // 0..15
  const int mr = mi * 32;
  const int nb = ni * 64;

  v8f acc[2][4];
#pragma unroll
  for (int i = 0; i < 2; ++i)
#pragma unroll
    for (int j = 0; j < 4; ++j) acc[i][j] = vzero8f();

  for (int kb = 0; kb < CH; kb += 32) {
    v16h a0 = load_A16x32(A, CH, mr,      kb, lane);
    v16h a1 = load_A16x32(A, CH, mr + 16, kb, lane);
    v16h b0 = load_B32x16(W, CH, nb,      kb, lane);
    v16h b1 = load_B32x16(W, CH, nb + 16, kb, lane);
    v16h b2 = load_B32x16(W, CH, nb + 32, kb, lane);
    v16h b3 = load_B32x16(W, CH, nb + 48, kb, lane);
    acc[0][0] = wmma_f32_f16(a0, b0, acc[0][0]);
    acc[0][1] = wmma_f32_f16(a0, b1, acc[0][1]);
    acc[0][2] = wmma_f32_f16(a0, b2, acc[0][2]);
    acc[0][3] = wmma_f32_f16(a0, b3, acc[0][3]);
    acc[1][0] = wmma_f32_f16(a1, b0, acc[1][0]);
    acc[1][1] = wmma_f32_f16(a1, b1, acc[1][1]);
    acc[1][2] = wmma_f32_f16(a1, b2, acc[1][2]);
    acc[1][3] = wmma_f32_f16(a1, b3, acc[1][3]);
  }

  const int rhalf = (lane >> 4) << 3;
#pragma unroll
  for (int i = 0; i < 2; ++i) {
#pragma unroll
    for (int j = 0; j < 4; ++j) {
      const int col = nb + 16 * j + (lane & 15);
#pragma unroll
      for (int r = 0; r < 8; ++r) {
        const int row = mr + 16 * i + r + rhalf;
        const size_t idx = (size_t)row * CH + col;
        const half_t mm = (half_t)acc[i][j][r];
        out[idx] = (half_t)(resid[idx] + mm);
      }
    }
  }
}

extern "C" void kernel_launch(void* const* d_in, const int* in_sizes, int n_in,
                              void* d_out, int out_size, void* d_ws, size_t ws_size,
                              hipStream_t stream) {
  (void)in_sizes; (void)n_in; (void)out_size; (void)ws_size;
  const half_t* residual = (const half_t*)d_in[0];
  const half_t* x        = (const half_t*)d_in[1];
  const half_t* w_qkv    = (const half_t*)d_in[2];
  const half_t* w_o      = (const half_t*)d_in[3];
  half_t* out = (half_t*)d_out;

  half_t* qkv  = (half_t*)d_ws;                                  // [3][B][H][T][D]
  half_t* attn = qkv + (size_t)3 * BATCH * NH * SEQ * HD;        // [B][T][C]

  // GEMM1: 8192x3072x1024 -> 256 M-tiles x 48 N-tiles, 8 waves/block.
  qkv_gemm_kernel<<<(256 * 48) / 8, 256, 0, stream>>>(x, w_qkv, qkv);
  // Flash attention: 64 (b,h) pairs x 16 query blocks of 128 rows.
  flash_attn_kernel<<<BATCH * NH * (SEQ / 128), 256, 0, stream>>>(qkv, attn);
  // GEMM2: 8192x1024x1024 + residual.
  out_proj_kernel<<<(256 * 16) / 8, 256, 0, stream>>>(attn, w_o, residual, out);
}
